// ControlVariate_23261542875179
// MI455X (gfx1250) — matrix operand order repaired and checked by reading, not compiled, not run on hardware
//
#include <hip/hip_runtime.h>
#include <math.h>

typedef __bf16 bf16_t;
typedef __attribute__((ext_vector_type(16))) __bf16 v16bf;
typedef __attribute__((ext_vector_type(8)))  float  v8f;

#define O_   64
#define PN_  64
#define H_   100
#define OBS_ 100
#define SD_  8
#define AD_  10
#define IN_  118
#define NNODES 63
#define GP_  112          // padded rows per gate
#define JP_  336          // 3 * GP_

// workspace byte offsets (all 32B-aligned)
#define OFF_WHH 0         // bf16 [336][128]  = 86016 B
#define OFF_WSA 86016     // bf16 [336][32]   = 21504 B
#define OFF_BHH 107520    // f32  [336]       = 1344 B (padded to 1536)
#define OFF_OI  109056    // f32  [64][336]   = 86016 B
#define OFF_C   195072    // f32  [4096]      = 16384 B

__device__ __forceinline__ v8f wmma_bf16(v16bf a, v16bf b, v8f c) {
  return __builtin_amdgcn_wmma_f32_16x16x32_bf16(false, a, false, b, (short)0, c, false, false);
}

// Pack W_hh -> bf16 [336][128] (gate-padded rows, K padded 100->128),
// W_ih sample+addr cols -> bf16 [336][32] (K padded 18->32), b_hh -> padded f32.
__global__ void prep_weights(const float* __restrict__ W_ih, const float* __restrict__ W_hh,
                             const float* __restrict__ b_hh,
                             bf16_t* __restrict__ Whh_p, bf16_t* __restrict__ Wsa_p,
                             float* __restrict__ bhh_p) {
  int jp = blockIdx.x;            // 0..335
  int k  = threadIdx.x;           // 0..127
  int g = jp / GP_, jj = jp % GP_;
  bool valid = jj < H_;
  int j = g * H_ + jj;            // real output-neuron row (if valid)
  float wh = (valid && k < H_) ? W_hh[j * H_ + k] : 0.f;
  Whh_p[jp * 128 + k] = (bf16_t)wh;
  if (k < 32) {
    float wsa = (valid && k < (SD_ + AD_)) ? W_ih[j * IN_ + OBS_ + k] : 0.f;
    Wsa_p[jp * 32 + k] = (bf16_t)wsa;
  }
  if (k == 0) bhh_p[jp] = valid ? b_hh[j] : 0.f;
}

// OI[o][jp] = b_ih[j] + obs[o] . W_ih[j, 0:100]   (f32, gate-padded)
__global__ void prep_oi(const float* __restrict__ obs, const float* __restrict__ W_ih,
                        const float* __restrict__ b_ih, float* __restrict__ OI_p) {
  int o = blockIdx.x;
  int jp = threadIdx.x;           // 0..335
  int g = jp / GP_, jj = jp % GP_;
  float acc = 0.f;
  if (jj < H_) {
    int j = g * H_ + jj;
    acc = b_ih[j];
    for (int k = 0; k < OBS_; ++k) acc += obs[o * OBS_ + k] * W_ih[j * IN_ + k];
  }
  OI_p[o * JP_ + jp] = acc;
}

// One workgroup = one (o,p) tree. 8 wave32 per block.
__global__ __launch_bounds__(256) void tree_kernel(
    const float* __restrict__ sample_emb, const float* __restrict__ addr_emb,
    const bf16_t* __restrict__ Whh_p, const bf16_t* __restrict__ Wsa_p,
    const float* __restrict__ bhh_p, const float* __restrict__ OI_p,
    const float* __restrict__ W1, const float* __restrict__ b1,
    const float* __restrict__ W2, const float* __restrict__ b2,
    const float* __restrict__ W3, const float* __restrict__ b3,
    float* __restrict__ cbuf) {
  __shared__ float hc[32 * 128];    // child hidden state, K-padded, zero rows >= w
  __shared__ float hout[32 * H_];   // this level's new hidden state
  const int tid  = threadIdx.x;
  const int lane = tid & 31;
  const int wave = tid >> 5;
  const int op = blockIdx.x;        // o*64 + p
  const int o  = op >> 6;

  for (int i = tid; i < 32 * 128; i += 256) hc[i] = 0.f;
  __syncthreads();

  const int m   = lane & 15;        // A row / B col within 16x16 tile
  const int hi  = lane >> 4;        // lane half
  const int kbA = hi * 16;          // A-fragment K base
  const int kbB = hi * 16;          // B-fragment K base

  for (int l = 5; l >= 0; --l) {
    const int w = 1 << l;
    const int s = w - 1;
    const int rt = (w + 15) >> 4;   // row tiles (2 for w=32, else 1)
    const int njobs = rt * 7;       // 7 col-tiles cover 112 padded cols per gate
    for (int job = wave; job < njobs; job += 8) {
      const int r  = job / 7;
      const int jt = job % 7;
      const int node = r * 16 + m;        // A-matrix row for this lane
      const bool nvalid = node < w;

      // ---- A fragment: sample+addr features (single K=32 step) ----
      v16bf a_sa;
      {
        const size_t base = (size_t)op * NNODES + (size_t)(s + (nvalid ? node : 0));
        const float* smp = sample_emb + base * SD_;
        const float* adr = addr_emb + base * AD_;
        union { v16bf v; bf16_t e[16]; } u;
#pragma unroll
        for (int i = 0; i < 16; ++i) {
          int kk = kbA + i;   // contiguous: kbA+i
          float f = 0.f;
          if (nvalid) {
            if (kk < SD_) f = smp[kk];
            else if (kk < SD_ + AD_) f = adr[kk - SD_];
          }
          u.e[i] = (bf16_t)f;
        }
        a_sa = u.v;
      }

      // ---- A fragments: child hidden state, 4 K-steps (K padded 100->128) ----
      v16bf a_h[4];
#pragma unroll
      for (int ks = 0; ks < 4; ++ks) {
        const float* pr = hc + node * 128 + ks * 32 + kbA;  // rows >= w are zero
        union { v16bf v; bf16_t e[16]; } u;
#pragma unroll
        for (int i = 0; i < 8; ++i) {
          u.e[i]     = (bf16_t)pr[i];
          u.e[8 + i] = (bf16_t)pr[8 + i];
        }
        a_h[ks] = u.v;
      }

      // ---- 3 gates: gi = SA-wmma + OI, gh = 4x hidden-wmma + b_hh ----
      float gi[3][8], gh[3][8];
#pragma unroll
      for (int g = 0; g < 3; ++g) {
        const int jp = g * GP_ + jt * 16 + m;   // padded weight row = B column
        v8f ai = {};
        {
          v16bf b = *(const v16bf*)(Wsa_p + (size_t)jp * 32 + kbB);
          ai = wmma_bf16(a_sa, b, ai);
        }
        v8f ah = {};
#pragma unroll
        for (int ks = 0; ks < 4; ++ks) {
          v16bf b = *(const v16bf*)(Whh_p + (size_t)jp * 128 + ks * 32 + kbB);
          ah = wmma_bf16(a_h[ks], b, ah);
        }
        const float oi = OI_p[o * JP_ + jp];
        const float bh = bhh_p[jp];
#pragma unroll
        for (int v = 0; v < 8; ++v) { gi[g][v] = ai[v] + oi; gh[g][v] = ah[v] + bh; }
      }

      // ---- GRU elementwise straight from accumulators ----
      const int c = jt * 16 + m;                // output column (0..111)
      if (c < H_) {
#pragma unroll
        for (int v = 0; v < 8; ++v) {
          const int nrow = r * 16 + hi * 8 + v; // D-matrix row
          if (nrow < w) {
            float hprev = hc[nrow * 128 + c];
            float rr = 1.f / (1.f + __expf(-(gi[0][v] + gh[0][v])));
            float zz = 1.f / (1.f + __expf(-(gi[1][v] + gh[1][v])));
            float nn = tanhf(gi[2][v] + rr * gh[2][v]);
            hout[nrow * H_ + c] = (1.f - zz) * nn + zz * hprev;
          }
        }
      }
    }
    __syncthreads();
    if (l > 0) {
      // child sum for next level: hc[b] = hout[2b] + hout[2b+1]; zero stale rows
      const int hw = w >> 1;
      for (int idx = tid; idx < hw * H_; idx += 256) {
        int b = idx / H_, cc = idx % H_;
        hc[b * 128 + cc] = hout[(2 * b) * H_ + cc] + hout[(2 * b + 1) * H_ + cc];
      }
      for (int idx = tid; idx < (w - hw) * H_; idx += 256) {
        int b = hw + idx / H_, cc = idx % H_;
        hc[b * 128 + cc] = 0.f;
      }
      __syncthreads();
    }
  }

  // ---- MLP on root = hout[0][0..99] (reuse hc as scratch) ----
  float* y1 = hc;
  float* y2 = hc + 64;
  if (tid < 50) {
    float a = b1[tid];
    for (int k = 0; k < H_; ++k) a += W1[tid * H_ + k] * hout[k];
    y1[tid] = fmaxf(a, 0.f);
  }
  __syncthreads();
  if (tid < 25) {
    float a = b2[tid];
    for (int k = 0; k < 50; ++k) a += W2[tid * 50 + k] * y1[k];
    y2[tid] = fmaxf(a, 0.f);
  }
  __syncthreads();
  if (tid == 0) {
    float a = b3[0];
    for (int k = 0; k < 25; ++k) a += W3[k] * y2[k];
    cbuf[op] = a;
  }
}

// logsumexp over particles, minus log(P)
__global__ void lse_kernel(const float* __restrict__ cbuf, float* __restrict__ out) {
  int o = threadIdx.x;
  if (o < O_) {
    float mx = -INFINITY;
    for (int p = 0; p < PN_; ++p) mx = fmaxf(mx, cbuf[o * PN_ + p]);
    float ss = 0.f;
    for (int p = 0; p < PN_; ++p) ss += __expf(cbuf[o * PN_ + p] - mx);
    out[o] = mx + logf(ss) - logf((float)PN_);
  }
}

extern "C" void kernel_launch(void* const* d_in, const int* in_sizes, int n_in,
                              void* d_out, int out_size, void* d_ws, size_t ws_size,
                              hipStream_t stream) {
  const float* obs  = (const float*)d_in[0];
  const float* samp = (const float*)d_in[1];
  const float* addr = (const float*)d_in[2];
  const float* W_ih = (const float*)d_in[3];
  const float* W_hh = (const float*)d_in[4];
  const float* b_ih = (const float*)d_in[5];
  const float* b_hh = (const float*)d_in[6];
  const float* W1   = (const float*)d_in[7];
  const float* b1   = (const float*)d_in[8];
  const float* W2   = (const float*)d_in[9];
  const float* b2   = (const float*)d_in[10];
  const float* W3   = (const float*)d_in[11];
  const float* b3   = (const float*)d_in[12];

  char* ws = (char*)d_ws;
  bf16_t* Whh_p = (bf16_t*)(ws + OFF_WHH);
  bf16_t* Wsa_p = (bf16_t*)(ws + OFF_WSA);
  float*  bhh_p = (float*)(ws + OFF_BHH);
  float*  OI_p  = (float*)(ws + OFF_OI);
  float*  cbuf  = (float*)(ws + OFF_C);

  prep_weights<<<JP_, 128, 0, stream>>>(W_ih, W_hh, b_hh, Whh_p, Wsa_p, bhh_p);
  prep_oi<<<O_, JP_, 0, stream>>>(obs, W_ih, b_ih, OI_p);
  tree_kernel<<<O_ * PN_, 256, 0, stream>>>(samp, addr, Whh_p, Wsa_p, bhh_p, OI_p,
                                            W1, b1, W2, b2, W3, b3, cbuf);
  lse_kernel<<<1, 64, 0, stream>>>(cbuf, (float*)d_out);
}
